// MultiHeadAttention_42657615184313
// MI455X (gfx1250) — matrix operand order repaired and checked by reading, not compiled, not run on hardware
//
#include <hip/hip_runtime.h>
#include <hip/hip_bf16.h>

// ---------------------------------------------------------------------------
// MI455X (gfx1250) multi-head attention: bf16 WMMA (f32 accumulate) everywhere,
// TDM (tensor_load_to_lds) double-buffered K/V staging in the flash kernel.
// H=768, NH=12, DK=64, B=2, S=4096.
// ---------------------------------------------------------------------------

typedef __attribute__((ext_vector_type(16))) __bf16   v16bf;
typedef __attribute__((ext_vector_type(8)))  float    v8f;
typedef __attribute__((ext_vector_type(4)))  unsigned u32x4;
typedef __attribute__((ext_vector_type(8)))  unsigned u32x8;

#define S_LEN   4096
#define H_DIM   768
#define NHEADS  12
#define DKH     64
#define BATCH   2
#define MROWS   (BATCH * S_LEN)        // 8192
#define MTILES  (MROWS / 16)           // 512
#define NTILES  (H_DIM / 16)           // 48

// ---- WMMA wrapper -----------------------------------------------------------
__device__ inline v8f wmma_bf16f32(v16bf a, v16bf b, v8f c) {
    // D = A(16x32 bf16) * B(32x16 bf16) + C(16x16 f32)
    return __builtin_amdgcn_wmma_f32_16x16x32_bf16(
        /*neg_a=*/false, a, /*neg_b=*/false, b,
        /*c_mod=*/(short)0, c, /*reuse_a=*/false, /*reuse_b=*/false);
}

// ---- Fragment loaders -------------------------------------------------------
// A-fragment (16x32, 16-bit): lane r=lane%16 holds row r; half=lane/16.
// Elements 0..7 -> K = k0 + half*8 + e ; 8..15 -> K = k0 + 16 + half*8 + (e-8).
__device__ inline v16bf load_a_f32src(const float* __restrict__ M, int ldm, int row0, int k0) {
    int lane = threadIdx.x & 31;
    int r = lane & 15, hf = lane >> 4;
    const float* p = M + (size_t)(row0 + r) * ldm + k0 + hf * 8;
    v16bf a;
#pragma unroll
    for (int e = 0; e < 8; ++e) a[e] = (__bf16)p[e];
#pragma unroll
    for (int e = 0; e < 8; ++e) a[8 + e] = (__bf16)p[16 + e];
    return a;
}

__device__ inline v16bf load_a_bf16src(const __bf16* M, int ldm, int row0, int k0) {
    int lane = threadIdx.x & 31;
    int r = lane & 15, hf = lane >> 4;
    const __bf16* p = M + (size_t)(row0 + r) * ldm + k0 + hf * 8;
    v16bf a;
#pragma unroll
    for (int e = 0; e < 8; ++e) a[e] = p[e];
#pragma unroll
    for (int e = 0; e < 8; ++e) a[8 + e] = p[16 + e];
    return a;
}

// B-fragment (32x16, 16-bit): lane n=lane%16 holds column n of B; half=lane/16.
// Elements 0..7 -> K = k0 + half*8 + e ; 8..15 -> K = k0 + 16 + half*8 + (e-8).
// Source "Bt" is row-major with row n == column n of B (i.e. B transposed).
__device__ inline v16bf load_b_bf16t(const __bf16* Bt, int ldb, int n0, int k0) {
    int lane = threadIdx.x & 31;
    int n = lane & 15, hf = lane >> 4;
    const __bf16* p = Bt + (size_t)(n0 + n) * ldb + k0 + hf * 8;
    v16bf b;
#pragma unroll
    for (int e = 0; e < 8; ++e) b[e] = p[e];
#pragma unroll
    for (int e = 0; e < 8; ++e) b[8 + e] = p[16 + e];
    return b;
}

// ---- TDM: 2D tile load Global -> LDS (D# per CDNA5 ISA) --------------
// data_size = 2 bytes (bf16). Wave-uniform arguments only.
__device__ inline void tdm_load_2d_bf16(unsigned lds_off_bytes, const void* gptr,
                                        unsigned tdim0, unsigned tdim1,
                                        unsigned tile0, unsigned tile1,
                                        unsigned stride0)
{
    unsigned long long ga = (unsigned long long)(size_t)gptr;
    u32x4 g0;
    g0[0] = 1u;                                            // count=1, user mode
    g0[1] = lds_off_bytes;                                 // lds_addr[31:0]
    g0[2] = (unsigned)(ga & 0xFFFFFFFFu);                  // global_addr[31:0]
    g0[3] = (unsigned)((ga >> 32) & 0x01FFFFFFu)           // global_addr[56:32]
          | 0x80000000u;                                   // type = 2 ("image")
    u32x8 g1;
    g1[0] = 0x00010000u;                                   // data_size=1 -> 2B
    g1[1] = (tdim0 & 0xFFFFu) << 16;                       // tensor_dim0[15:0]
    g1[2] = (tdim0 >> 16) | ((tdim1 & 0xFFFFu) << 16);     // dim0[31:16]|dim1[15:0]
    g1[3] = (tdim1 >> 16) | (tile0 << 16);                 // dim1[31:16]|tile_dim0
    g1[4] = tile1;                                         // tile_dim1 (tile_dim2=0)
    g1[5] = stride0;                                       // tensor_dim0_stride[31:0]
    g1[6] = 0u;                                            // stride0[47:32]|stride1[15:0]
    g1[7] = 0u;                                            // stride1[47:16]
    asm volatile("tensor_load_to_lds %0, %1" :: "s"(g0), "s"(g1) : "memory");
}

// ---- cross-lane helpers (within each 16-lane half of the wave32) -----------
__device__ inline float rowmax16(float x) {
    x = fmaxf(x, __shfl_xor(x, 1, 32));
    x = fmaxf(x, __shfl_xor(x, 2, 32));
    x = fmaxf(x, __shfl_xor(x, 4, 32));
    x = fmaxf(x, __shfl_xor(x, 8, 32));
    return x;
}
__device__ inline float rowsum16(float x) {
    x += __shfl_xor(x, 1, 32);
    x += __shfl_xor(x, 2, 32);
    x += __shfl_xor(x, 4, 32);
    x += __shfl_xor(x, 8, 32);
    return x;
}

// ===========================================================================
// Kernel 1: convert + transpose the four weight matrices to bf16.
// Wt[m][n][k] = (bf16) W_m[k][n]  (m = q,k,v,o)
// ===========================================================================
__global__ void __launch_bounds__(256) mha_wconv_kernel(
    const float* __restrict__ Wq, const float* __restrict__ Wk,
    const float* __restrict__ Wv, const float* __restrict__ Wo,
    __bf16* __restrict__ Wt)
{
    int idx = blockIdx.x * 256 + threadIdx.x;          // over 4*768*768
    int m = idx / (H_DIM * H_DIM);
    int r = idx % (H_DIM * H_DIM);
    int n = r / H_DIM;
    int k = r % H_DIM;
    const float* W = (m == 0) ? Wq : (m == 1) ? Wk : (m == 2) ? Wv : Wo;
    Wt[idx] = (__bf16)W[(size_t)k * H_DIM + n];
}

// ===========================================================================
// Kernel 2: fused QKV projection.  One wave per 16x16 output tile.
//   Q,K -> bf16 [B,NH,S,DK] ;  V -> bf16 transposed [B,NH,DK,S]
// ===========================================================================
__global__ void __launch_bounds__(256) mha_qkv_proj_kernel(
    const float* __restrict__ Xq, const float* __restrict__ Xk, const float* __restrict__ Xv,
    const __bf16* __restrict__ Wt,      // [3][768][768] transposed bf16 (q,k,v)
    const float* __restrict__ bq, const float* __restrict__ bk, const float* __restrict__ bv,
    __bf16* __restrict__ Qh, __bf16* __restrict__ Kh, __bf16* __restrict__ Vt)
{
    int t = blockIdx.x * 8 + (threadIdx.x >> 5);       // wave-tile id
    int mat = t / (MTILES * NTILES);
    int r   = t % (MTILES * NTILES);
    int mt = r / NTILES, nt = r % NTILES;
    const float*  X    = (mat == 0) ? Xq : (mat == 1) ? Xk : Xv;
    const __bf16* W    = Wt + (size_t)mat * H_DIM * H_DIM;
    const float*  bias = (mat == 0) ? bq : (mat == 1) ? bk : bv;

    int m0 = mt * 16, n0 = nt * 16;
    v8f acc = {};
#pragma unroll 4
    for (int k0 = 0; k0 < H_DIM; k0 += 32) {
        v16bf a = load_a_f32src(X, H_DIM, m0, k0);
        v16bf b = load_b_bf16t(W, H_DIM, n0, k0);
        acc = wmma_bf16f32(a, b, acc);
    }

    int lane = threadIdx.x & 31;
    int nn = lane & 15, hf = lane >> 4;
    int col = n0 + nn;
    int h = col >> 6, dk = col & 63;
    float bcol = bias[col];
#pragma unroll
    for (int v = 0; v < 8; ++v) {
        int m = m0 + hf * 8 + v;
        int bb = m >> 12;                    // batch
        int s  = m & (S_LEN - 1);
        __bf16 val = (__bf16)(acc[v] + bcol);
        size_t bh = (size_t)(bb * NHEADS + h);
        if (mat == 0)      Qh[(bh * S_LEN + s) * DKH + dk] = val;
        else if (mat == 1) Kh[(bh * S_LEN + s) * DKH + dk] = val;
        else               Vt[(bh * DKH + dk) * S_LEN + s] = val;
    }
}

// ===========================================================================
// Kernel 3: cooperative flash attention.
//   One block (8 waves) per (b,h, 128-row q-block); wave w owns q-tile w.
//   K tile (32x64) and V tile (64x32) staged in LDS by the Tensor Data Mover,
//   double-buffered: DMA of tile i+1 overlaps WMMA work on tile i.
// Dynamic LDS layout (bytes):
//   [0,8192)      : K tiles, 2 buffers of 32x64 bf16 (4 KB each)
//   [8192,16384)  : V tiles, 2 buffers of 64x32 bf16 (4 KB each)
//   [16384,24576) : per-wave P relayout tiles, 8 x (16x32 bf16)
// ===========================================================================
__global__ void __launch_bounds__(256) mha_flash_kernel(
    const __bf16* __restrict__ Qh, const __bf16* __restrict__ Kh,
    const __bf16* __restrict__ Vt, const int* __restrict__ mask,
    float* __restrict__ ctx)
{
    extern __shared__ char smem[];
    int widx = threadIdx.x >> 5;
    int bh = blockIdx.x >> 5;                          // 0..23  (b*NH + h)
    int qb = blockIdx.x & 31;                          // 0..31  q-block of 128
    int b  = bh / NHEADS, h = bh % NHEADS;

    const __bf16* Qb = Qh + (size_t)bh * S_LEN * DKH;
    const __bf16* Kb = Kh + (size_t)bh * S_LEN * DKH;
    const __bf16* Vb = Vt + (size_t)bh * DKH * S_LEN;
    const int*    Mb = mask + (size_t)b * S_LEN * S_LEN;

    int q0 = qb * 128 + widx * 16;
    int lane = threadIdx.x & 31;
    int nn = lane & 15, hf = lane >> 4;

    // Q tile resident in registers (16 x 64 bf16 -> two A fragments)
    v16bf qa0 = load_a_bf16src(Qb, DKH, q0, 0);
    v16bf qa1 = load_a_bf16src(Qb, DKH, q0, 32);

    v8f o0 = {}, o1 = {}, o2 = {}, o3 = {};
    float mrow[8], lrow[8];
#pragma unroll
    for (int v = 0; v < 8; ++v) { mrow[v] = -1.0e30f; lrow[v] = 0.0f; }

    __bf16* myp = (__bf16*)(smem + 16384) + widx * 512;

    // prologue: DMA first K/V tiles into buffer 0 (wave 0 only; TDM ignores EXEC)
    if (widx == 0) {
        tdm_load_2d_bf16(0,    Kb, DKH,   S_LEN, DKH, 32,  DKH);    // K: 32 rows x 64
        tdm_load_2d_bf16(8192, Vb, S_LEN, DKH,   32,  DKH, S_LEN);  // V: 64 rows x 32
    }

    for (int kb = 0; kb < S_LEN; kb += 32) {
        int cur = (kb >> 5) & 1;
        __syncthreads();   // previous compute done -> safe to overwrite buf cur^1
        if (widx == 0) {
            int kn = kb + 32;
            if (kn < S_LEN) {
                int nb = cur ^ 1;
                tdm_load_2d_bf16((unsigned)(nb * 4096), Kb + (size_t)kn * DKH,
                                 DKH, S_LEN, DKH, 32, DKH);
                tdm_load_2d_bf16((unsigned)(8192 + nb * 4096), Vb + kn,
                                 S_LEN, DKH, 32, DKH, S_LEN);
                __builtin_amdgcn_s_wait_tensorcnt(2);   // current tiles landed
            } else {
                __builtin_amdgcn_s_wait_tensorcnt(0);
            }
        }
        __syncthreads();   // LDS tiles for this iteration visible to all waves

        const __bf16* Kt = (const __bf16*)(smem + cur * 4096);          // [32][64]
        const __bf16* Vs = (const __bf16*)(smem + 8192 + cur * 4096);   // [64][32]

        // --- scores: two 16x16 C tiles over local keys [0,16) and [16,32)
        v8f s0 = {}, s1 = {};
        s0 = wmma_bf16f32(qa0, load_b_bf16t(Kt, DKH, 0,  0),  s0);
        s0 = wmma_bf16f32(qa1, load_b_bf16t(Kt, DKH, 0,  32), s0);
        s1 = wmma_bf16f32(qa0, load_b_bf16t(Kt, DKH, 16, 0),  s1);
        s1 = wmma_bf16f32(qa1, load_b_bf16t(Kt, DKH, 16, 32), s1);

        // --- scale, mask, online softmax (row stats live in C layout)
#pragma unroll
        for (int v = 0; v < 8; ++v) {
            int qrow = q0 + hf * 8 + v;
            const int* mr = Mb + (size_t)qrow * S_LEN + kb;
            float x0 = s0[v] * 0.125f; if (mr[nn]      == 0) x0 = -1.0e9f;
            float x1 = s1[v] * 0.125f; if (mr[16 + nn] == 0) x1 = -1.0e9f;
            float bm   = rowmax16(fmaxf(x0, x1));
            float mnew = fmaxf(mrow[v], bm);
            float corr = __expf(mrow[v] - mnew);
            float p0 = __expf(x0 - mnew);
            float p1 = __expf(x1 - mnew);
            float ps = rowsum16(p0 + p1);
            lrow[v] = lrow[v] * corr + ps;
            mrow[v] = mnew;
            o0[v] *= corr; o1[v] *= corr; o2[v] *= corr; o3[v] *= corr;
            int rr = hf * 8 + v;
            myp[rr * 32 + nn]      = (__bf16)p0;   // P tile row-major 16x32
            myp[rr * 32 + 16 + nn] = (__bf16)p1;
        }

        // --- P (C layout) -> A fragment via per-wave LDS tile (same-wave DS
        //     ops are in order; compiler inserts s_wait_dscnt)
        v16bf pa = load_a_bf16src(myp, 32, 0, 0);

        // --- O += P(16x32) * V(32x16) per DK tile; V tile rows contiguous
        o0 = wmma_bf16f32(pa, load_b_bf16t(Vs, 32, 0,  0), o0);
        o1 = wmma_bf16f32(pa, load_b_bf16t(Vs, 32, 16, 0), o1);
        o2 = wmma_bf16f32(pa, load_b_bf16t(Vs, 32, 32, 0), o2);
        o3 = wmma_bf16f32(pa, load_b_bf16t(Vs, 32, 48, 0), o3);
    }

    // --- normalize rows and write merged-head ctx [B,S,H] fp32
#pragma unroll
    for (int v = 0; v < 8; ++v) {
        float inv = 1.0f / lrow[v];
        int s = q0 + hf * 8 + v;
        float* dst = ctx + ((size_t)b * S_LEN + s) * H_DIM + h * DKH;
        dst[nn]      = o0[v] * inv;
        dst[16 + nn] = o1[v] * inv;
        dst[32 + nn] = o2[v] * inv;
        dst[48 + nn] = o3[v] * inv;
    }
}

// ===========================================================================
// Kernel 4: output projection  out = ctx @ Wo + bo   (fp32 out)
// ===========================================================================
__global__ void __launch_bounds__(256) mha_out_proj_kernel(
    const float* __restrict__ ctx, const __bf16* __restrict__ Wot,
    const float* __restrict__ bo, float* __restrict__ out)
{
    int t = blockIdx.x * 8 + (threadIdx.x >> 5);
    int mt = t / NTILES, nt = t % NTILES;
    int m0 = mt * 16, n0 = nt * 16;
    v8f acc = {};
#pragma unroll 4
    for (int k0 = 0; k0 < H_DIM; k0 += 32) {
        v16bf a = load_a_f32src(ctx, H_DIM, m0, k0);
        v16bf b = load_b_bf16t(Wot, H_DIM, n0, k0);
        acc = wmma_bf16f32(a, b, acc);
    }
    int lane = threadIdx.x & 31;
    int nn = lane & 15, hf = lane >> 4;
    float bcol = bo[n0 + nn];
#pragma unroll
    for (int v = 0; v < 8; ++v) {
        int m = m0 + hf * 8 + v;
        out[(size_t)m * H_DIM + n0 + nn] = acc[v] + bcol;
    }
}

// ===========================================================================
extern "C" void kernel_launch(void* const* d_in, const int* in_sizes, int n_in,
                              void* d_out, int out_size, void* d_ws, size_t ws_size,
                              hipStream_t stream) {
    const float* q    = (const float*)d_in[0];
    const float* k    = (const float*)d_in[1];
    const float* v    = (const float*)d_in[2];
    const int*   mask = (const int*)  d_in[3];
    const float* Wq   = (const float*)d_in[4];
    const float* bq   = (const float*)d_in[5];
    const float* Wk   = (const float*)d_in[6];
    const float* bk   = (const float*)d_in[7];
    const float* Wv   = (const float*)d_in[8];
    const float* bv   = (const float*)d_in[9];
    const float* Wo   = (const float*)d_in[10];
    const float* bo   = (const float*)d_in[11];
    float* out = (float*)d_out;

    // workspace layout
    char* ws = (char*)d_ws;
    const size_t WT_BYTES   = (size_t)4 * H_DIM * H_DIM * 2;            //  4.7 MB
    const size_t HEAD_BYTES = (size_t)BATCH * NHEADS * S_LEN * DKH * 2; // 12.6 MB
    __bf16* Wt  = (__bf16*)ws;
    __bf16* Qh  = (__bf16*)(ws + WT_BYTES);
    __bf16* Kh  = (__bf16*)(ws + WT_BYTES + HEAD_BYTES);
    __bf16* Vtp = (__bf16*)(ws + WT_BYTES + 2 * HEAD_BYTES);
    float*  ctx = (float*) (ws + WT_BYTES + 3 * HEAD_BYTES);            // 25.2 MB

    // 1) weight convert+transpose: 4*768*768 elems
    mha_wconv_kernel<<<(4 * H_DIM * H_DIM) / 256, 256, 0, stream>>>(Wq, Wk, Wv, Wo, Wt);

    // 2) fused QKV projection: 3*512*48 wave-tiles, 8 waves/block
    mha_qkv_proj_kernel<<<(3 * MTILES * NTILES) / 8, 256, 0, stream>>>(
        q, k, v, Wt, bq, bk, bv, Qh, Kh, Vtp);

    // 3) cooperative flash attention: 24 head-batches * 32 q-blocks = 768 blocks
    mha_flash_kernel<<<BATCH * NHEADS * (S_LEN / 128), 256, 24576, stream>>>(
        Qh, Kh, Vtp, mask, ctx);

    // 4) output projection: 512*48 wave-tiles
    mha_out_proj_kernel<<<(MTILES * NTILES) / 8, 256, 0, stream>>>(
        ctx, Wt + (size_t)3 * H_DIM * H_DIM, bo, out);
}
